// PointPWC_51719996178541
// MI455X (gfx1250) — matrix operand order repaired and checked by reading, not compiled, not run on hardware
//
#include <hip/hip_runtime.h>

typedef __attribute__((ext_vector_type(16))) _Float16 v16h;
typedef __attribute__((ext_vector_type(8)))  _Float16 v8h;
typedef __attribute__((ext_vector_type(8)))  float    v8f;

constexpr int NPTS = 4096;   // N1 == N2
constexpr int KNB  = 32;     // nsample
constexpr int CH   = 64;     // channels
constexpr int CIN  = 131;    // 64 + 64 + 3
constexpr int CINP = 160;    // padded to multiple of 32 for f16 WMMA K-steps

// ---------------------------------------------------------------------------
// Kernel 1: brute-force KNN. One thread per query point; reference points
// tiled through LDS; top-32 kept in registers via replace-max (unrolled
// select chains keep the arrays in VGPRs, no scratch).
// ---------------------------------------------------------------------------
__launch_bounds__(128)
__global__ void knn_kernel(const float* __restrict__ qxyz,  // [B,3,N]
                           const float* __restrict__ rxyz,  // [B,3,N]
                           int* __restrict__ out_idx)       // [B,N,32]
{
    __shared__ float srx[128], sry[128], srz[128];
    const int gid = blockIdx.x * 128 + threadIdx.x;   // b*NPTS + n
    const int b = gid / NPTS;
    const int n = gid - b * NPTS;
    const float* qb = qxyz + (size_t)b * 3 * NPTS;
    const float* rb = rxyz + (size_t)b * 3 * NPTS;
    const float qx = qb[n], qy = qb[NPTS + n], qz = qb[2 * NPTS + n];

    float bd[KNB];
    int   bi[KNB];
#pragma unroll
    for (int s = 0; s < KNB; ++s) { bd[s] = 3.0e38f; bi[s] = 0; }
    float maxv = 3.0e38f;
    int   maxp = 0;

    for (int t0 = 0; t0 < NPTS; t0 += 128) {
        __syncthreads();
        const int li = threadIdx.x;
        srx[li] = rb[t0 + li];
        sry[li] = rb[NPTS + t0 + li];
        srz[li] = rb[2 * NPTS + t0 + li];
        __syncthreads();
        for (int j = 0; j < 128; ++j) {
            const float dx = qx - srx[j], dy = qy - sry[j], dz = qz - srz[j];
            const float d = dx * dx + dy * dy + dz * dz;
            if (d < maxv) {
                const int ii = t0 + j;
#pragma unroll
                for (int s = 0; s < KNB; ++s)
                    if (s == maxp) { bd[s] = d; bi[s] = ii; }
                maxv = -1.0f;
#pragma unroll
                for (int s = 0; s < KNB; ++s)
                    if (bd[s] > maxv) { maxv = bd[s]; maxp = s; }
            }
        }
    }
#pragma unroll
    for (int s = 0; s < KNB; ++s)
        out_idx[(size_t)gid * KNB + s] = bi[s];
}

// ---------------------------------------------------------------------------
// Kernel 2: one-time f32 -> f16 weight conversion with K-padding.
// ---------------------------------------------------------------------------
__global__ void prep_weights(const float* __restrict__ w0,  // [64,131]
                             const float* __restrict__ w1,  // [64,64]
                             _Float16* __restrict__ w0h,    // [64,160]
                             _Float16* __restrict__ w1h)    // [64,64]
{
    const int stride = gridDim.x * blockDim.x;
    int t = blockIdx.x * blockDim.x + threadIdx.x;
    for (int i = t; i < CH * CINP; i += stride) {
        const int o = i / CINP, c = i - o * CINP;
        w0h[i] = (c < CIN) ? (_Float16)w0[o * CIN + c] : (_Float16)0.0f;
    }
    for (int i = t; i < CH * CH; i += stride)
        w1h[i] = (_Float16)w1[i];
}

// ---------------------------------------------------------------------------
// Kernel 3: stage 1. Gather + concat rows, 131->64->64 MLP via
// v_wmma_f32_16x16x32_f16, weightnet1, weighted reduction over K.
// Block = 128 threads (4 waves) handles 2 query points = 64 rows.
// ---------------------------------------------------------------------------
__launch_bounds__(128)
__global__ void costvol_mlp_kernel(
    const float* __restrict__ xyz1, const float* __restrict__ xyz2,
    const float* __restrict__ pts1, const float* __restrict__ pts2,
    const _Float16* __restrict__ w0h, const _Float16* __restrict__ w1h,
    const float* __restrict__ b0,  const float* __restrict__ b1,
    const float* __restrict__ wn_w0, const float* __restrict__ wn_b0,
    const float* __restrict__ wn_w1, const float* __restrict__ wn_b1,
    const float* __restrict__ wn_w2, const float* __restrict__ wn_b2,
    const int* __restrict__ idx1,
    float* __restrict__ p2p)                    // [B,N,64]
{
    // Manual LDS arena, 60160 bytes (< 64KB static limit).
    __shared__ alignas(64) unsigned char smem[60160];
    _Float16* sW0 = (_Float16*)(smem);           // [64][160]  20480 B
    _Float16* sW1 = (_Float16*)(smem + 20480);   // [64][64]    8192 B
    _Float16* sX  = (_Float16*)(smem + 28672);   // [64][160]  20480 B
    float*    sH1 = (float*)   (smem + 28672);   // [64][64]   aliases sX (safe: barrier-separated)
    _Float16* sH  = (_Float16*)(smem + 49152);   // [64][64]    8192 B
    float*    sH8 = (float*)   (smem + 57344);   // [64][8]     2048 B
    float*    sD  = (float*)   (smem + 59392);   // [64][3]      768 B

    const int tid = threadIdx.x;
    const int blk = blockIdx.x;
    const int b   = blk / (NPTS / 2);
    const int n0  = (blk - b * (NPTS / 2)) * 2;

    // ---- load weights into LDS ----
    for (int i = tid; i < CH * CINP; i += 128) sW0[i] = w0h[i];
    for (int i = tid; i < CH * CH;   i += 128) sW1[i] = w1h[i];

    // ---- stage input rows: [gp1(64) | gp2(64) | dir(3) | zeros] as f16 ----
    {
        const int row  = tid >> 1;
        const int half = tid & 1;
        const int q = row >> 5, k = row & 31;
        const int n = n0 + q;
        const int i = idx1[((size_t)(b * NPTS + n)) * KNB + k];
        const float* p1b = pts1 + (size_t)b * CH * NPTS;
        const float* p2b = pts2 + (size_t)b * CH * NPTS;
        _Float16* xr = sX + row * CINP;
        if (half == 0) {
            for (int c = 0; c < 64; ++c)  xr[c] = (_Float16)p1b[c * NPTS + n];
            for (int c = 64; c < 80; ++c) xr[c] = (_Float16)p2b[(c - 64) * NPTS + i];
        } else {
            for (int c = 80; c < 128; ++c) xr[c] = (_Float16)p2b[(c - 64) * NPTS + i];
            const float* x1 = xyz1 + (size_t)b * 3 * NPTS;
            const float* x2 = xyz2 + (size_t)b * 3 * NPTS;
            const float d0 = x2[i] - x1[n];
            const float d1 = x2[NPTS + i] - x1[NPTS + n];
            const float d2 = x2[2 * NPTS + i] - x1[2 * NPTS + n];
            xr[128] = (_Float16)d0; xr[129] = (_Float16)d1; xr[130] = (_Float16)d2;
            for (int c = CIN; c < CINP; ++c) xr[c] = (_Float16)0.0f;
            sD[row * 3 + 0] = d0; sD[row * 3 + 1] = d1; sD[row * 3 + 2] = d2;
        }
    }
    __syncthreads();

    const int wv    = tid >> 5;          // wave id: owns 16-row stripe
    const int l     = tid & 31;
    const int lrow  = l & 15;
    const int khalf = (l < 16) ? 0 : 16; // contiguous k order (matches B)
    const int kb    = (l < 16) ? 0 : 16; // contiguous k order (matches A)
    const int R     = wv * 16;

    // ---- layer 0: [64x160] x [160x64] via WMMA ----
#pragma unroll
    for (int t = 0; t < 4; ++t) {
        v8f acc = {};
#pragma unroll
        for (int c = 0; c < 5; ++c) {
            const _Float16* ar = sX + (R + lrow) * CINP + c * 32;
            v8h lo = *(const v8h*)(ar + khalf);
            v8h hi = *(const v8h*)(ar + 8 + khalf);
            v16h a;
#pragma unroll
            for (int e = 0; e < 8; ++e) { a[e] = lo[e]; a[e + 8] = hi[e]; }
            v16h bb = *(const v16h*)(sW0 + (t * 16 + lrow) * CINP + c * 32 + kb);
            acc = __builtin_amdgcn_wmma_f32_16x16x32_f16(
                false, a, false, bb, (short)0, acc, false, false);
        }
        const int col = t * 16 + lrow;
        const float bias = b0[col];
#pragma unroll
        for (int e = 0; e < 8; ++e) {
            float v = acc[e] + bias;
            v = (v >= 0.0f) ? v : 0.1f * v;                 // leaky
            sH[(R + e + ((l < 16) ? 0 : 8)) * CH + col] = (_Float16)v;
        }
    }
    __syncthreads();

    // ---- layer 1: [64x64] x [64x64] via WMMA ----
#pragma unroll
    for (int t = 0; t < 4; ++t) {
        v8f acc = {};
#pragma unroll
        for (int c = 0; c < 2; ++c) {
            const _Float16* ar = sH + (R + lrow) * CH + c * 32;
            v8h lo = *(const v8h*)(ar + khalf);
            v8h hi = *(const v8h*)(ar + 8 + khalf);
            v16h a;
#pragma unroll
            for (int e = 0; e < 8; ++e) { a[e] = lo[e]; a[e + 8] = hi[e]; }
            v16h bb = *(const v16h*)(sW1 + (t * 16 + lrow) * CH + c * 32 + kb);
            acc = __builtin_amdgcn_wmma_f32_16x16x32_f16(
                false, a, false, bb, (short)0, acc, false, false);
        }
        const int col = t * 16 + lrow;
        const float bias = b1[col];
#pragma unroll
        for (int e = 0; e < 8; ++e) {
            float v = acc[e] + bias;
            v = (v >= 0.0f) ? v : 0.1f * v;                 // leaky
            sH1[(R + e + ((l < 16) ? 0 : 8)) * CH + col] = v;
        }
    }
    __syncthreads();

    // ---- weightnet1 hidden (3->8->8, ReLU), one row per thread ----
    if (tid < 64) {
        const float d0 = sD[tid * 3], d1 = sD[tid * 3 + 1], d2 = sD[tid * 3 + 2];
        float ha[8];
#pragma unroll
        for (int h = 0; h < 8; ++h) {
            float v = wn_w0[h * 3] * d0 + wn_w0[h * 3 + 1] * d1
                    + wn_w0[h * 3 + 2] * d2 + wn_b0[h];
            ha[h] = v > 0.0f ? v : 0.0f;
        }
#pragma unroll
        for (int h = 0; h < 8; ++h) {
            float v = wn_b1[h];
#pragma unroll
            for (int j = 0; j < 8; ++j) v += wn_w1[h * 8 + j] * ha[j];
            sH8[tid * 8 + h] = v > 0.0f ? v : 0.0f;
        }
    }
    __syncthreads();

    // ---- p2p_cost[b,n,c] = sum_k relu(w2 . h8 + b2) * h1 ----
    {
        const int q = tid >> 6, c = tid & 63;
        float wc[8];
#pragma unroll
        for (int j = 0; j < 8; ++j) wc[j] = wn_w2[c * 8 + j];
        const float bc = wn_b2[c];
        float acc = 0.0f;
#pragma unroll
        for (int k = 0; k < KNB; ++k) {
            const int row = q * KNB + k;
            float w = bc;
#pragma unroll
            for (int j = 0; j < 8; ++j) w += wc[j] * sH8[row * 8 + j];
            w = w > 0.0f ? w : 0.0f;
            acc += w * sH1[row * CH + c];
        }
        p2p[((size_t)(b * NPTS) + n0 + q) * CH + c] = acc;
    }
}

// ---------------------------------------------------------------------------
// Kernel 4: stage 2. Gather p2p_cost at self-KNN indices, weightnet2,
// weighted reduction, transposed store [B,64,N].
// ---------------------------------------------------------------------------
__launch_bounds__(256)
__global__ void patch2patch_kernel(
    const float* __restrict__ xyz1,
    const float* __restrict__ p2p,              // [B,N,64]
    const int* __restrict__ idx2,               // [B,N,32]
    const float* __restrict__ wn_w0, const float* __restrict__ wn_b0,
    const float* __restrict__ wn_w1, const float* __restrict__ wn_b1,
    const float* __restrict__ wn_w2, const float* __restrict__ wn_b2,
    float* __restrict__ out)                    // [B,64,N]
{
    __shared__ float sH8[128][8];
    __shared__ int   sIdx[128];
    const int tid = threadIdx.x;
    const int blk = blockIdx.x;
    const int b   = blk / (NPTS / 4);
    const int n0  = (blk - b * (NPTS / 4)) * 4;

    if (tid < 128) {
        const int row = tid, q = row >> 5, k = row & 31;
        const int n = n0 + q;
        const int i = idx2[((size_t)(b * NPTS + n)) * KNB + k];
        sIdx[row] = i;
        const float* x1 = xyz1 + (size_t)b * 3 * NPTS;
        const float d0 = x1[i] - x1[n];
        const float d1 = x1[NPTS + i] - x1[NPTS + n];
        const float d2 = x1[2 * NPTS + i] - x1[2 * NPTS + n];
        float ha[8];
#pragma unroll
        for (int h = 0; h < 8; ++h) {
            float v = wn_w0[h * 3] * d0 + wn_w0[h * 3 + 1] * d1
                    + wn_w0[h * 3 + 2] * d2 + wn_b0[h];
            ha[h] = v > 0.0f ? v : 0.0f;
        }
#pragma unroll
        for (int h = 0; h < 8; ++h) {
            float v = wn_b1[h];
#pragma unroll
            for (int j = 0; j < 8; ++j) v += wn_w1[h * 8 + j] * ha[j];
            sH8[row][h] = v > 0.0f ? v : 0.0f;
        }
    }
    __syncthreads();

    {
        const int q = tid >> 6, c = tid & 63;
        const int n = n0 + q;
        float wc[8];
#pragma unroll
        for (int j = 0; j < 8; ++j) wc[j] = wn_w2[c * 8 + j];
        const float bc = wn_b2[c];
        const float* pb = p2p + (size_t)b * NPTS * CH;
        float acc = 0.0f;
#pragma unroll
        for (int k = 0; k < KNB; ++k) {
            const int row = q * KNB + k;
            float w = bc;
#pragma unroll
            for (int j = 0; j < 8; ++j) w += wc[j] * sH8[row][j];
            w = w > 0.0f ? w : 0.0f;
            acc += w * pb[(size_t)sIdx[row] * CH + c];
        }
        out[((size_t)b * CH + c) * NPTS + n] = acc;
    }
}

// ---------------------------------------------------------------------------
// Host launcher
// ---------------------------------------------------------------------------
extern "C" void kernel_launch(void* const* d_in, const int* in_sizes, int n_in,
                              void* d_out, int out_size, void* d_ws, size_t ws_size,
                              hipStream_t stream) {
    const float* xyz1    = (const float*)d_in[0];
    const float* xyz2    = (const float*)d_in[1];
    const float* points1 = (const float*)d_in[2];
    const float* points2 = (const float*)d_in[3];
    const float* w_mlp0  = (const float*)d_in[4];
    const float* b_mlp0  = (const float*)d_in[5];
    const float* w_mlp1  = (const float*)d_in[6];
    const float* b_mlp1  = (const float*)d_in[7];
    const float* wn1_w0  = (const float*)d_in[8];
    const float* wn1_b0  = (const float*)d_in[9];
    const float* wn1_w1  = (const float*)d_in[10];
    const float* wn1_b1  = (const float*)d_in[11];
    const float* wn1_w2  = (const float*)d_in[12];
    const float* wn1_b2  = (const float*)d_in[13];
    const float* wn2_w0  = (const float*)d_in[14];
    const float* wn2_b0  = (const float*)d_in[15];
    const float* wn2_w1  = (const float*)d_in[16];
    const float* wn2_b1  = (const float*)d_in[17];
    const float* wn2_w2  = (const float*)d_in[18];
    const float* wn2_b2  = (const float*)d_in[19];

    constexpr int B = 2;
    unsigned char* ws = (unsigned char*)d_ws;
    int*       idx1 = (int*)(ws);                       // 1 MB
    int*       idx2 = (int*)(ws + (1u << 20));          // 1 MB
    float*     p2p  = (float*)(ws + (2u << 20));        // 2 MB
    _Float16*  w0h  = (_Float16*)(ws + (4u << 20));     // 20480 B
    _Float16*  w1h  = (_Float16*)(ws + (4u << 20) + 32768); // 8192 B
    float*     out  = (float*)d_out;

    // 1) KNN: p1 -> p2 and p1 -> p1
    knn_kernel<<<(B * NPTS) / 128, 128, 0, stream>>>(xyz1, xyz2, idx1);
    knn_kernel<<<(B * NPTS) / 128, 128, 0, stream>>>(xyz1, xyz1, idx2);

    // 2) weight conversion (f32 -> padded f16)
    prep_weights<<<48, 256, 0, stream>>>(w_mlp0, w_mlp1, w0h, w1h);

    // 3) stage-1 cost volume: WMMA MLP + weightnet1 + K-reduction
    costvol_mlp_kernel<<<(B * NPTS) / 2, 128, 0, stream>>>(
        xyz1, xyz2, points1, points2, w0h, w1h, b_mlp0, b_mlp1,
        wn1_w0, wn1_b0, wn1_w1, wn1_b1, wn1_w2, wn1_b2, idx1, p2p);

    // 4) stage-2 patch-to-patch
    patch2patch_kernel<<<(B * NPTS) / 4, 256, 0, stream>>>(
        xyz1, p2p, idx2, wn2_w0, wn2_b0, wn2_w1, wn2_b1, wn2_w2, wn2_b2, out);
}
